// _UniDirectionalStack_67473936220933
// MI455X (gfx1250) — compile-verified, matrix operand order checked by reading, not run on hardware
//
#include <hip/hip_runtime.h>
#include <hip/hip_bf16.h>

// ---------------------------------------------------------------------------
// CDNA5 (gfx1250) implicit-GEMM ConvLSTM stack.
// Conv gates computed with v_wmma_f32_16x16x32_bf16 (bf16 in, f32 accum).
// Block tile: 256 gates x 32 pixels; each wave32: 2 M-tiles x 2 N-tiles.
// Separate LDS regions for patch / B-fragments / gate tile so no phase needs
// to execute before a barrier it doesn't depend on (keeps the GEMM loop
// free of hoisted loads -> no scratch spills, VGPRs < 256).
// ---------------------------------------------------------------------------

typedef __attribute__((ext_vector_type(16))) __bf16 v16bf;
typedef __attribute__((ext_vector_type(8)))  float  v8f;

union Frag { uint4 u[2]; v16bf v; };

__device__ __forceinline__ unsigned short f2bf(float f) {
    unsigned int u = __float_as_uint(f);
    u += 0x7FFFu + ((u >> 16) & 1u);          // round-to-nearest-even
    return (unsigned short)(u >> 16);
}
__device__ __forceinline__ float bf2f(unsigned short s) {
    return __uint_as_float(((unsigned int)s) << 16);
}
__device__ __forceinline__ float sigm(float x) { return 1.f / (1.f + __expf(-x)); }

// ---------------------------------------------------------------------------
// Pre-swizzle fp32 conv weights [256, cinTot, 3, 3] into bf16 A-fragments.
// Layout: aw[((mtile*nkt + ktile)*32 + lane)*16 + j], j follows the ISA
// 16-bit A-matrix 16x32 lane layout: K = (j>>3)*16 + (lane>>4)*8 + (j&7),
// M = mtile*16 + (lane&15).  K-order inside the GEMM: k = (ky*3+kx)*cinTot+cin.
// ---------------------------------------------------------------------------
__global__ void prep_weights_kernel(const float* __restrict__ gw,
                                    unsigned short* __restrict__ aw,
                                    int cinTot, int nkt) {
    int total = nkt * 16 * 512;
    for (int e = blockIdx.x * blockDim.x + threadIdx.x; e < total;
         e += gridDim.x * blockDim.x) {
        int j = e & 15, lane = (e >> 4) & 31, tile = e >> 9;
        int kt = tile % nkt, mt = tile / nkt;
        int m = (mt << 4) + (lane & 15);
        int k = (kt << 5) + ((j >> 3) << 4) + ((lane >> 4) << 3) + (j & 7);
        int rpos = k / cinTot;             // (ky*3+kx)
        int cin  = k - rpos * cinTot;
        int ky = rpos / 3, kx = rpos - ky * 3;
        aw[e] = f2bf(gw[((m * cinTot + cin) * 3 + ky) * 3 + kx]);
    }
}

// ---------------------------------------------------------------------------
// One ConvLSTM step for one layer (all batches, all pixels).
// Block = 256 threads (8 wave32).  Block tile: 256 gates x 32 pixels.
// Wave w: M-tiles {2w, 2w+1} x N-tiles {0,1}; A loaded once per K-tile,
// B fragments shared across all 8 waves through LDS.
// ---------------------------------------------------------------------------
template <int CIN_X, int DIL, bool RESP>
__global__ __launch_bounds__(256)
void lstm_step(const float* __restrict__ xt, long long bsx,
               const float* __restrict__ hp, long long bsh,
               const unsigned short* __restrict__ aw,
               const float* __restrict__ gb,
               const float* __restrict__ rw,
               const float* __restrict__ rb,
               float* __restrict__ cbuf,
               float* __restrict__ ho, long long bso,
               int t) {
    constexpr int CINT   = CIN_X + 64;            // concat channels
    constexpr int NKT    = (CINT * 9) / 32;       // K-tiles of 32
    constexpr int PW     = 32 + 2 * DIL;          // staged patch width
    constexpr int PATCHE = CINT * 3 * PW;         // patch elements (bf16)

    __shared__ __align__(16) unsigned short patch[PATCHE];    // input tile (bf16)
    __shared__ __align__(16) unsigned short bfr[NKT * 1024];  // B frags, 2 N-tiles
    __shared__ __align__(16) float          gt[256 * 32];     // gate tile (f32)
    __shared__ __align__(16) float          rwS[RESP ? 64 * 32 : 16];

    const int tid = threadIdx.x;
    const int b   = blockIdx.x >> 7;      // 128 N-tiles of 32 pixels per batch
    const int nt  = blockIdx.x & 127;
    const int y   = nt >> 1;
    const int x0  = (nt & 1) << 5;

    // ---- stage dilated input patch (concat(x_t, h_prev)) as bf16 ----
    for (int e = tid; e < PATCHE; e += 256) {
        int cx = e % PW; int rr = e / PW; int ry = rr % 3; int cin = rr / 3;
        int yy = y + DIL * (ry - 1);
        int xx = x0 + cx - DIL;
        float v = 0.f;
        if (yy >= 0 && yy < 64 && xx >= 0 && xx < 64) {
            if (cin < CIN_X)
                v = xt[b * bsx + (long long)cin * 4096 + yy * 64 + xx];
            else if (t > 0)
                v = hp[b * bsh + (long long)(cin - CIN_X) * 4096 + yy * 64 + xx];
        }
        patch[e] = f2bf(v);
    }
    if (RESP) {                           // stage 1x1 projection weights (8 KB)
        for (int e = tid; e < 64 * 32; e += 256) rwS[e] = rw[e];
    }
    __syncthreads();

    // ---- build all B fragments once per block (shared by 8 waves) ----
    // bfr[((kt*2 + sub)*32 + lane)*16 + j], sub = N-tile half.
    for (int e = tid; e < NKT * 1024; e += 256) {
        int j = e & 15, lane = (e >> 4) & 31, sub = (e >> 9) & 1, kt = e >> 10;
        int n = (sub << 4) + (lane & 15);
        int k = (kt << 5) + ((j >> 3) << 4) + ((lane >> 4) << 3) + (j & 7);
        int rpos = k / CINT; int cin = k - rpos * CINT;
        int ky = rpos / 3, kx = rpos - ky * 3;
        bfr[e] = patch[(cin * 3 + ky) * PW + n + kx * DIL];
    }
    __syncthreads();

    // ---- WMMA GEMM: gates[256 x 32] = W[256 x K] * im2col[K x 32] ----
    const int wave = tid >> 5, lane = tid & 31;
    const int mt0 = wave * 2, mt1 = mt0 + 1;
    v8f acc00 = {0,0,0,0,0,0,0,0}, acc01 = {0,0,0,0,0,0,0,0};
    v8f acc10 = {0,0,0,0,0,0,0,0}, acc11 = {0,0,0,0,0,0,0,0};
    const unsigned short* a0p = aw + (((size_t)mt0 * NKT) << 9) + (lane << 4);
    const unsigned short* a1p = aw + (((size_t)mt1 * NKT) << 9) + (lane << 4);
    const unsigned short* bp  = bfr + (lane << 4);
#pragma clang loop unroll_count(2)
    for (int kt = 0; kt < NKT; ++kt) {
        Frag fa0, fa1, fb0, fb1;
        const uint4* p0  = (const uint4*)a0p;
        const uint4* p1  = (const uint4*)a1p;
        const uint4* pb0 = (const uint4*)bp;
        const uint4* pb1 = (const uint4*)(bp + 512);
        fa0.u[0] = p0[0];  fa0.u[1] = p0[1];
        fa1.u[0] = p1[0];  fa1.u[1] = p1[1];
        fb0.u[0] = pb0[0]; fb0.u[1] = pb0[1];
        fb1.u[0] = pb1[0]; fb1.u[1] = pb1[1];
        a0p += 512; a1p += 512; bp += 1024;
        acc00 = __builtin_amdgcn_wmma_f32_16x16x32_bf16(false, fa0.v, false, fb0.v,
                                                        (short)0, acc00, false, false);
        acc01 = __builtin_amdgcn_wmma_f32_16x16x32_bf16(false, fa0.v, false, fb1.v,
                                                        (short)0, acc01, false, false);
        acc10 = __builtin_amdgcn_wmma_f32_16x16x32_bf16(false, fa1.v, false, fb0.v,
                                                        (short)0, acc10, false, false);
        acc11 = __builtin_amdgcn_wmma_f32_16x16x32_bf16(false, fa1.v, false, fb1.v,
                                                        (short)0, acc11, false, false);
    }

    // ---- spill accumulators via the documented C/D lane layout ----
    {
        int nn = lane & 15, mr = (lane >> 4) << 3;   // lanes 16-31 hold M=8..15
#pragma unroll
        for (int r = 0; r < 8; r++) {
            gt[((mt0 << 4) + mr + r) * 32 + nn]      = acc00[r];
            gt[((mt0 << 4) + mr + r) * 32 + 16 + nn] = acc01[r];
            gt[((mt1 << 4) + mr + r) * 32 + nn]      = acc10[r];
            gt[((mt1 << 4) + mr + r) * 32 + 16 + nn] = acc11[r];
        }
    }
    __syncthreads();

    // ---- LSTM pointwise + residual:
    //      c' = s(f)c + s(i)tanh(g); h = s(o)tanh(c') + res ----
#pragma unroll
    for (int q = 0; q < 8; q++) {
        int p = tid + (q << 8); int hid = p >> 5; int n = p & 31;
        float r;
        if (RESP) {               // layer 0: 1x1 projection of x_t (from LDS)
            r = rb[hid];
            for (int cin = 0; cin < CIN_X; cin++)
                r += rwS[hid * CIN_X + cin] * bf2f(patch[(cin * 3 + 1) * PW + n + DIL]);
        } else {                  // identity residual
            r = xt[b * bsx + (long long)hid * 4096 + y * 64 + x0 + n];
        }
        float ig = gt[hid * 32 + n]         + gb[hid];
        float fg = gt[(64  + hid) * 32 + n] + gb[64  + hid];
        float gg = gt[(128 + hid) * 32 + n] + gb[128 + hid];
        float og = gt[(192 + hid) * 32 + n] + gb[192 + hid];
        long long cix = (long long)b * (64 * 4096) + (long long)hid * 4096 + y * 64 + x0 + n;
        float cp = (t > 0) ? cbuf[cix] : 0.f;
        float cn = sigm(fg) * cp + sigm(ig) * tanhf(gg);
        float hn = sigm(og) * tanhf(cn) + r;
        cbuf[cix] = cn;
        ho[b * bso + (long long)hid * 4096 + y * 64 + x0 + n] = hn;
    }
}

// ---------------------------------------------------------------------------
extern "C" void kernel_launch(void* const* d_in, const int* in_sizes, int n_in,
                              void* d_out, int out_size, void* d_ws, size_t ws_size,
                              hipStream_t stream) {
    (void)in_sizes; (void)n_in; (void)out_size; (void)ws_size;
    const float* x   = (const float*)d_in[0];
    const float* gw0 = (const float*)d_in[1];
    const float* gb0 = (const float*)d_in[2];
    const float* gw1 = (const float*)d_in[3];
    const float* gb1 = (const float*)d_in[4];
    const float* gw2 = (const float*)d_in[5];
    const float* gb2 = (const float*)d_in[6];
    const float* rw0 = (const float*)d_in[7];
    const float* rb0 = (const float*)d_in[8];

    unsigned char* ws = (unsigned char*)d_ws;
    size_t off = 0;
    auto take = [&](size_t bytes) -> unsigned char* {
        unsigned char* p = ws + off;
        off = (off + bytes + 255) & ~(size_t)255;
        return p;
    };
    unsigned short* aw0 = (unsigned short*)take((size_t)27 * 16 * 512 * 2);
    unsigned short* aw1 = (unsigned short*)take((size_t)36 * 16 * 512 * 2);
    unsigned short* aw2 = (unsigned short*)take((size_t)36 * 16 * 512 * 2);
    float* seq0 = (float*)take((size_t)4 * 16 * 64 * 4096 * 4);
    float* seq1 = (float*)take((size_t)4 * 16 * 64 * 4096 * 4);
    float* cbuf = (float*)take((size_t)4 * 64 * 4096 * 4);

    // Pre-swizzle weights into WMMA A-fragment order (bf16).
    prep_weights_kernel<<<256, 256, 0, stream>>>(gw0, aw0, 96,  27);
    prep_weights_kernel<<<256, 256, 0, stream>>>(gw1, aw1, 128, 36);
    prep_weights_kernel<<<256, 256, 0, stream>>>(gw2, aw2, 128, 36);

    const long long HW = 4096;
    // Layer 0: Cin_x=32, dil=1, projected residual.
    for (int t = 0; t < 16; t++) {
        const float* xtp = x    + (long long)t * 32 * HW;
        const float* hpp = seq0 + (long long)(t > 0 ? t - 1 : 0) * 64 * HW;
        float*       hop = seq0 + (long long)t * 64 * HW;
        lstm_step<32, 1, true><<<512, 256, 0, stream>>>(
            xtp, 16 * 32 * HW, hpp, 16 * 64 * HW,
            aw0, gb0, rw0, rb0, cbuf, hop, 16 * 64 * HW, t);
    }
    // Layer 1: Cin_x=64, dil=2, identity residual.
    for (int t = 0; t < 16; t++) {
        const float* xtp = seq0 + (long long)t * 64 * HW;
        const float* hpp = seq1 + (long long)(t > 0 ? t - 1 : 0) * 64 * HW;
        float*       hop = seq1 + (long long)t * 64 * HW;
        lstm_step<64, 2, false><<<512, 256, 0, stream>>>(
            xtp, 16 * 64 * HW, hpp, 16 * 64 * HW,
            aw1, gb1, nullptr, nullptr, cbuf, hop, 16 * 64 * HW, t);
    }
    // Layer 2: Cin_x=64, dil=4, identity residual; writes d_out directly.
    float* outp = (float*)d_out;
    for (int t = 0; t < 16; t++) {
        const float* xtp = seq1 + (long long)t * 64 * HW;
        const float* hpp = outp + (long long)(t > 0 ? t - 1 : 0) * 64 * HW;
        float*       hop = outp + (long long)t * 64 * HW;
        lstm_step<64, 4, false><<<512, 256, 0, stream>>>(
            xtp, 16 * 64 * HW, hpp, 16 * 64 * HW,
            aw2, gb2, nullptr, nullptr, cbuf, hop, 16 * 64 * HW, t);
    }
}
